// Correlation_919123001767
// MI455X (gfx1250) — matrix operand order repaired and checked by reading, not compiled, not run on hardware
//
#include <hip/hip_runtime.h>

// ---------------------------------------------------------------------------
// PWC-Net correlation on MI455X (gfx1250).
//
// Roofline: 11.3 GFLOP of channel dot products vs ~323 MB HBM traffic
// (~14 us at 23.3 TB/s). f16 WMMA (v_wmma_f32_16x16x32_f16, K=C=128 in 4
// k-steps) makes compute sub-dominant, so the design minimizes data movement:
//   1) one-time f32 NCHW -> parity-deinterleaved padded NHWC f16 (97 MB total,
//      L2-resident at 192 MB),
//   2) one fused kernel: banded all-pairs WMMA in u-space (dx even => same
//      parity => contiguous du band, only 3 N-tiles per M-tile) -> LDS ->
//      3x3 box sum -> normalized output [4,441,156,156].
// ---------------------------------------------------------------------------

typedef __attribute__((ext_vector_type(16))) _Float16 v16h;
typedef __attribute__((ext_vector_type(8)))  _Float16 v8h;
typedef __attribute__((ext_vector_type(8)))  float    v8f;

#define Bn     4
#define Cc     128
#define Hh     192
#define Ww     192
#define PADc   3
#define HP     198       // H + 2*PAD
#define UW     120       // u-columns per parity plane incl. margins
#define USHIFT 8         // stored upc = (padcol>>1) + USHIFT
#define UOFF   18        // s1-u -> stored index: upc = u + 10 + USHIFT
#define Dd     21
#define DDd    441
#define HOUT   156

// --------------------------------------------------------------------------
// Prep: f32 NCHW -> f16 parity-deinterleaved padded NHWC.
// dst[b][r][q][upc][c], r = h+PAD, q = (w+PAD)&1, upc = ((w+PAD)>>1)+USHIFT.
// Margins pre-zeroed by hipMemsetAsync. LDS 32x32 tile for coalescing.
// --------------------------------------------------------------------------
__global__ __launch_bounds__(256)
void prep_nhwc_f16(const float* __restrict__ src, _Float16* __restrict__ dst) {
  __shared__ float tile[32][33];
  const int tx = threadIdx.x, ty = threadIdx.y;
  const int w0 = blockIdx.x * 32;
  const int h  = blockIdx.y;
  const int z  = blockIdx.z;        // b*4 + (channel tile)
  const int b  = z >> 2;
  const int c0 = (z & 3) * 32;
#pragma unroll
  for (int j = 0; j < 4; ++j) {
    int c = c0 + ty + j * 8;
    tile[ty + j * 8][tx] =
        src[(((size_t)b * Cc + c) * Hh + h) * Ww + (w0 + tx)];
  }
  __syncthreads();
#pragma unroll
  for (int j = 0; j < 4; ++j) {
    int wl = ty + j * 8;
    int pc = w0 + wl + PADc;                   // padded column
    size_t o = ((((size_t)b * HP + (h + PADc)) * 2 + (pc & 1)) * UW +
                ((pc >> 1) + USHIFT)) * Cc + (c0 + tx);
    dst[o] = (_Float16)tile[tx][wl];
  }
}

// --------------------------------------------------------------------------
// Main fused correlation kernel.
// Block = (b, dy index, tile of 8 output rows). 8 waves, 256 threads.
// Jobs: 10 f1 rows x 10 M-tiles (2 parities x 5 u-tiles). Per job: load A
// fragments (4 k-steps), then 3 N-tiles x 4 chained WMMAs; scatter the du in
// [-10,10] band of each 16x16 tile into LDS prod[y][dx_idx][x]; then 3x3 box
// sum + normalize + store.
// --------------------------------------------------------------------------
__global__ __launch_bounds__(256)
void corr_wmma(const _Float16* __restrict__ p1,
               const _Float16* __restrict__ p2,
               float* __restrict__ out) {
  __shared__ float prod[10 * 21 * 160];        // [y_local][dx_idx][x], 134 KB

  const int ho0  = blockIdx.x * 8;             // output-row tile base
  const int dyi  = blockIdx.y;                 // 0..20
  const int b    = blockIdx.z;
  const int dy   = (dyi - 10) * 2;             // -20..20 step 2
  const int lane = threadIdx.x & 31;
  const int wave = threadIdx.x >> 5;
  const int lo   = lane & 15;
  const bool hi  = lane >= 16;

  const size_t plane = (size_t)HP * 2 * UW * Cc;
  const _Float16* base1 = p1 + (size_t)b * plane;
  const _Float16* base2 = p2 + (size_t)b * plane;

  for (int job = wave; job < 100; job += 8) {
    const int yl = job / 10;                   // f1 row within tile (0..9)
    const int mt = job % 10;
    const int p  = mt & 1;                     // column parity plane
    const int u0 = (mt >> 1) * 16;             // M-tile base in u-space
    const int y1 = ho0 + yl;                   // s1 row (0..161)
    const int r1 = y1 + 20;                    // pad1 row, <= 181 < 198
    int r2 = y1 + 20 + dy;                     // pad2 row
    if (r2 > HP - 1) r2 = HP - 1;              // clamp tail garbage rows

    // ---- A fragments: 16x32 f16, M = u position, K = channel ----
    // lanes 0-15: row M=lane, halves 0-15 = K c0+0..15
    // lanes 16-31: same row, K offset +16.
    const _Float16* rowA =
        base1 + (((size_t)r1 * 2 + p) * UW + (u0 + lo + UOFF)) * Cc;
    v16h afr[4];
#pragma unroll
    for (int ks = 0; ks < 4; ++ks) {
      const int cb = ks * 32 + (hi ? 16 : 0);
      v8h a0 = *(const v8h*)(rowA + cb);
      v8h a1 = *(const v8h*)(rowA + cb + 8);
      v16h a;
#pragma unroll
      for (int i = 0; i < 8; ++i) { a[i] = a0[i]; a[8 + i] = a1[i]; }
      afr[ks] = a;
    }

    const _Float16* rowB = base2 + ((size_t)r2 * 2 + p) * UW * Cc;
    const int xbase = 2 * u0 + p;              // s1 x of M row m=0
#pragma unroll
    for (int t = 0; t < 3; ++t) {
      const int u1 = u0 - 16 + 16 * t;         // N-tile base in u-space
      // B fragment: 32x16 f16, column N = lane; lanes 0-15 hold K 0..15,
      // lanes 16-31 hold K 16..31 (contiguous 16 halves = 32B per lane).
      const _Float16* colB =
          rowB + (size_t)(u1 + lo + UOFF) * Cc + (hi ? 16 : 0);
      v8f acc = {};
#pragma unroll
      for (int ks = 0; ks < 4; ++ks) {
        v16h bfr = *(const v16h*)(colB + ks * 32);
        acc = __builtin_amdgcn_wmma_f32_16x16x32_f16(
            /*neg_a=*/false, afr[ks], /*neg_b=*/false, bfr,
            /*c_mod=*/(short)0, acc, /*reuse_a=*/false, /*reuse_b=*/false);
      }
      // Scatter band entries (|du| <= 10) to LDS. C/D layout: VGPR v,
      // lanes 0-15 -> (M=v, N=lane); lanes 16-31 -> (M=v+8, N=lane-16).
      // Element index is affine in m: idx(m) = lbase - 158*m, valid iff
      // 0 <= du0 - m <= 20. Each (x,dx) pair is written exactly once.
      const int du0   = (u1 + lo) - u0 + 10;   // dx index at m = 0
      const int lbase = (yl * 21 + du0) * 160 + xbase;
#pragma unroll
      for (int v = 0; v < 8; ++v) {
        const int m = v + (hi ? 8 : 0);
        if ((unsigned)(du0 - m) <= 20u) {
          prod[lbase - 158 * m] = acc[v];
        }
      }
    }
  }
  __syncthreads();

  // ---- Phase 2: 3x3 box sum + normalize + store ----
  const float inv = 1.0f / 1152.0f;            // 1 / (K*K*C)
  for (int idx = threadIdx.x; idx < 8 * 21 * 156; idx += 256) {
    const int wo  = idx % 156;
    const int r   = idx / 156;
    const int dxi = r % 21;
    const int hl  = r / 21;
    const int ho  = ho0 + hl;
    if (ho >= HOUT) continue;
    float s = 0.0f;
#pragma unroll
    for (int ky = 0; ky < 3; ++ky)
#pragma unroll
      for (int kx = 0; kx < 3; ++kx)
        s += prod[((hl + ky) * 21 + dxi) * 160 + (wo + kx)];
    out[(((size_t)b * DDd + (dyi * 21 + dxi)) * HOUT + ho) * HOUT + wo] =
        s * inv;
  }
}

// --------------------------------------------------------------------------
extern "C" void kernel_launch(void* const* d_in, const int* in_sizes, int n_in,
                              void* d_out, int out_size, void* d_ws,
                              size_t ws_size, hipStream_t stream) {
  const float* in1 = (const float*)d_in[0];
  const float* in2 = (const float*)d_in[1];
  float* out = (float*)d_out;

  const size_t PBUF_ELEMS = (size_t)Bn * HP * 2 * UW * Cc;  // 24,330,240
  _Float16* p1 = (_Float16*)d_ws;
  _Float16* p2 = p1 + PBUF_ELEMS;

  // Zero both padded f16 buffers (margins must read as 0). Capturable.
  hipMemsetAsync(d_ws, 0, 2 * PBUF_ELEMS * sizeof(_Float16), stream);

  dim3 tb(32, 8, 1);
  dim3 gprep(Ww / 32, Hh, Bn * (Cc / 32));
  prep_nhwc_f16<<<gprep, tb, 0, stream>>>(in1, p1);
  prep_nhwc_f16<<<gprep, tb, 0, stream>>>(in2, p2);

  dim3 gmain(20, Dd, Bn);                      // (ho tiles, dy, batch)
  corr_wmma<<<gmain, 256, 0, stream>>>(p1, p2, out);
}
